// I2I_CrossAttentionWithGateFusion_8340826489179
// MI455X (gfx1250) — matrix operand order repaired and checked by reading, not compiled, not run on hardware
//
#include <hip/hip_runtime.h>
#include <hip/hip_bf16.h>

typedef _Float16 v16h __attribute__((ext_vector_type(16)));
typedef _Float16 v8h  __attribute__((ext_vector_type(8)));
typedef float    v8f  __attribute__((ext_vector_type(8)));

#define NTOK   8192       // D*H*W
#define CCH    128
#define NHEAD  4
#define DH     32
#define BK     64         // keys per flash-attention iteration
// scale * log2(e): baked into Q so softmax runs in base-2 (native v_exp_f32)
#define QSCALE_LOG2E (0.17677669529663687f * 1.4426950408889634f)

// ---------------------------------------------------------------------------
// WMMA helpers (gfx1250, wave32): D[16x16] = A[16x32]*B[32x16] + C (f16->f32)
// ---------------------------------------------------------------------------
__device__ __forceinline__ v8f wmma_f16(v16h a, v16h b, v8f c) {
    return __builtin_amdgcn_wmma_f32_16x16x32_f16(
        false, a, false, b, (short)0, c, false, false);
}

__device__ __forceinline__ v8f vzero8() {
    v8f z;
#pragma unroll
    for (int i = 0; i < 8; ++i) z[i] = 0.0f;
    return z;
}

// A fragment: 16x32 tile, source row-major [row][K] with stride ld halves.
// Lane L: row L&15; halves 0..7 = K kh..kh+7, halves 8..15 = K kh+16..kh+23,
// kh = 8*(L>>4).
__device__ __forceinline__ v16h load_frag_a(const _Float16* __restrict__ p,
                                            int ld, int kbase) {
    int lane = threadIdx.x & 31;
    int row  = lane & 15;
    int kh   = (lane >> 4) * 8;
    const _Float16* q = p + (size_t)row * ld + kbase + kh;
    union { v16h v; v8h h[2]; } f;
    f.h[0] = *reinterpret_cast<const v8h*>(q);
    f.h[1] = *reinterpret_cast<const v8h*>(q + 16);
    return f.v;
}

// B fragment: 32x16 tile, source per-column [col][K] with stride ld halves.
// Lane L: column L&15; halves 0..7 = K kh..kh+7, halves 8..15 = K kh+16..kh+23,
// kh = 8*(L>>4).
__device__ __forceinline__ v16h load_frag_b(const _Float16* __restrict__ p,
                                            int ld, int kbase) {
    int lane = threadIdx.x & 31;
    int col  = lane & 15;
    int kh   = (lane >> 4) * 8;
    const _Float16* q = p + (size_t)col * ld + kbase + kh;
    union { v16h v; v8h h[2]; } f;
    f.h[0] = *reinterpret_cast<const v8h*>(q);
    f.h[1] = *reinterpret_cast<const v8h*>(q + 16);
    return f.v;
}

// ---------------------------------------------------------------------------
// Async Global->LDS copy (CDNA5, ASYNCcnt-tracked).  Per ISA §10.2 the low 32
// bits of a flat pointer into LDS are the LDS byte address, so the VDST
// (LDS-address) VGPR is just the truncated generic pointer.
// ---------------------------------------------------------------------------
__device__ __forceinline__ void async_ld_b128(_Float16* lds_dst,
                                              const _Float16* gsrc) {
    unsigned loff = (unsigned)(unsigned long long)lds_dst;
    asm volatile("global_load_async_to_lds_b128 %0, %1, off"
                 :: "v"(loff), "v"(gsrc) : "memory");
}
#define WAIT_ASYNCCNT_LE4() asm volatile("s_wait_asynccnt 0x4" ::: "memory")

// ---------------------------------------------------------------------------
// Prep kernels
// ---------------------------------------------------------------------------
__global__ void transpose_to_f16(const float* __restrict__ src,
                                 _Float16* __restrict__ dst) {
    int idx = blockIdx.x * blockDim.x + threadIdx.x;   // over N*C
    if (idx >= NTOK * CCH) return;
    int n = idx / CCH, c = idx % CCH;
    dst[idx] = (_Float16)src[(size_t)c * NTOK + n];    // dst[n][c]
}

__global__ void cast_to_f16(const float* __restrict__ src,
                            _Float16* __restrict__ dst, int n) {
    int idx = blockIdx.x * blockDim.x + threadIdx.x;
    if (idx < n) dst[idx] = (_Float16)src[idx];
}

// ---------------------------------------------------------------------------
// QKV projection: out^T[n,o] = sum_c featT[n,c]*W[o,c] + b[o]
// blockIdx.y: 0=Q(feat1, scaled by QSCALE_LOG2E), 1=K(feat2), 2=V(feat2)
// q,k stored [H][N][DH]; v stored [H][DH][N].
// ---------------------------------------------------------------------------
__global__ __launch_bounds__(128) void proj_kernel(
    const _Float16* __restrict__ f1T, const _Float16* __restrict__ f2T,
    const _Float16* __restrict__ Wq16, const _Float16* __restrict__ Wk16,
    const _Float16* __restrict__ Wv16,
    const float* __restrict__ bq, const float* __restrict__ bk,
    const float* __restrict__ bv,
    _Float16* __restrict__ qT, _Float16* __restrict__ kT,
    _Float16* __restrict__ vT) {
    int which = blockIdx.y;
    const _Float16* A = (which == 0) ? f1T : f2T;
    const _Float16* W = (which == 0) ? Wq16 : (which == 1 ? Wk16 : Wv16);
    const float* bias = (which == 0) ? bq : (which == 1 ? bk : bv);

    int wave = threadIdx.x >> 5, lane = threadIdx.x & 31;
    int nbase = blockIdx.x * 64 + wave * 16;

    v8f acc[8];
#pragma unroll
    for (int t = 0; t < 8; ++t) acc[t] = vzero8();

#pragma unroll
    for (int kc = 0; kc < CCH; kc += 32) {
        v16h a = load_frag_a(A + (size_t)nbase * CCH, CCH, kc);
#pragma unroll
        for (int ct = 0; ct < 8; ++ct) {
            v16h b = load_frag_b(W + (size_t)ct * 16 * CCH, CCH, kc);
            acc[ct] = wmma_f16(a, b, acc[ct]);
        }
    }

    int hi = lane >> 4, cl = lane & 15;
#pragma unroll
    for (int ct = 0; ct < 8; ++ct) {
#pragma unroll
        for (int r = 0; r < 8; ++r) {
            int o = ct * 16 + cl;
            int n = nbase + r + 8 * hi;
            float val = acc[ct][r] + bias[o];
            if (which == 0) val *= QSCALE_LOG2E;
            int h = o >> 5, d = o & 31;
            if (which == 2) {
                vT[(size_t)h * DH * NTOK + (size_t)d * NTOK + n] = (_Float16)val;
            } else {
                _Float16* dst = (which == 1) ? kT : qT;
                dst[(size_t)h * NTOK * DH + (size_t)n * DH + d] = (_Float16)val;
            }
        }
    }
}

// ---------------------------------------------------------------------------
// Flash attention. Grid (N/64, NHEAD), 4 waves x 16 queries. 128 iterations
// of 64 keys. K/V tiles double-buffered in LDS via async global->LDS copies
// (each of the 128 threads issues exactly 4 b128 copies per iteration, so
// s_wait_asynccnt<=4 + barrier guarantees the current buffer landed).
// ---------------------------------------------------------------------------
__global__ __launch_bounds__(128) void attn_kernel(
    const _Float16* __restrict__ qT, const _Float16* __restrict__ kT,
    const _Float16* __restrict__ vT, _Float16* __restrict__ outT) {
    __shared__ __align__(16) _Float16 ldsK[2][BK * DH];   // [key][ch] 4KB
    __shared__ __align__(16) _Float16 ldsV[2][DH * BK];   // [ch][key] 4KB
    __shared__ __align__(16) _Float16 ldsP[4][16 * BK];   // per-wave P 2KB

    int tid = threadIdx.x;
    int wave = tid >> 5, lane = tid & 31;
    int h = blockIdx.y;
    int nbase = blockIdx.x * 64 + wave * 16;
    int hi = lane >> 4, cl = lane & 15;

    const _Float16* qh = qT + (size_t)h * NTOK * DH;
    const _Float16* kh = kT + (size_t)h * NTOK * DH;
    const _Float16* vh = vT + (size_t)h * DH * NTOK;

    v16h aq = load_frag_a(qh + (size_t)nbase * DH, DH, 0);

    v8f o0 = vzero8(), o1 = vzero8();
    float mrow[8], lrow[8];
#pragma unroll
    for (int r = 0; r < 8; ++r) { mrow[r] = -1e30f; lrow[r] = 0.0f; }

    _Float16* pw = ldsP[wave];

    // stage tile helper: 4 async b128 copies per thread
    auto stage = [&](int buf, int mb) {
#pragma unroll
        for (int j = 0; j < 2; ++j) {           // K tile: 4KB contiguous
            int chunk = tid + j * 128;          // 0..255, 16B each
            async_ld_b128(&ldsK[buf][chunk * 8], kh + (size_t)mb * DH + chunk * 8);
        }
#pragma unroll
        for (int j = 0; j < 2; ++j) {           // V tile: 32 rows x 128B
            int e = tid + j * 128;
            int c = e >> 3, x = (e & 7) * 8;
            async_ld_b128(&ldsV[buf][c * BK + x], vh + (size_t)c * NTOK + mb + x);
        }
    };

    stage(0, 0);

    for (int it = 0; it < NTOK / BK; ++it) {
        int cur = it & 1, nxt = cur ^ 1;
        int mbn = (it + 1) * BK;
        if (mbn >= NTOK) mbn = 0;               // dummy (uniform issue count)
        stage(nxt, mbn);
        WAIT_ASYNCCNT_LE4();                    // current buffer complete
        __syncthreads();                        // visible to all waves

        // --- scores: 4 tiles of 16 keys, K=DH=32 fully reduced ----------
        v8f s[4];
#pragma unroll
        for (int t = 0; t < 4; ++t) {
            v16h bk = load_frag_b(&ldsK[cur][t * 16 * DH], DH, 0);
            s[t] = wmma_f16(aq, bk, vzero8());
        }

        // --- online softmax over 64 keys (base-2 domain) ----------------
        float alpha[8];
#pragma unroll
        for (int r = 0; r < 8; ++r) {
            float mx = fmaxf(fmaxf(s[0][r], s[1][r]), fmaxf(s[2][r], s[3][r]));
#pragma unroll
            for (int off = 8; off; off >>= 1)
                mx = fmaxf(mx, __shfl_xor(mx, off, 32));
            float mnew = fmaxf(mrow[r], mx);
            float al = exp2f(mrow[r] - mnew);
            mrow[r] = mnew;
            alpha[r] = al;
            float rs = 0.0f;
#pragma unroll
            for (int t = 0; t < 4; ++t) {
                float p = exp2f(s[t][r] - mnew);
                s[t][r] = p;
                rs += p;
            }
#pragma unroll
            for (int off = 8; off; off >>= 1)
                rs += __shfl_xor(rs, off, 32);
            lrow[r] = lrow[r] * al + rs;
        }

        // --- P (C/D layout) -> LDS [row][key] for A-fragment reuse ------
#pragma unroll
        for (int t = 0; t < 4; ++t)
#pragma unroll
            for (int r = 0; r < 8; ++r)
                pw[(r + 8 * hi) * BK + t * 16 + cl] = (_Float16)s[t][r];

        // --- O += P * V  (K=64 in 2 chunks, 2 channel tiles) ------------
        v16h ap0 = load_frag_a(pw, BK, 0);
        v16h ap1 = load_frag_a(pw, BK, 32);
#pragma unroll
        for (int r = 0; r < 8; ++r) { o0[r] *= alpha[r]; o1[r] *= alpha[r]; }
        {
            v16h bv = load_frag_b(&ldsV[cur][0], BK, 0);
            o0 = wmma_f16(ap0, bv, o0);
            bv = load_frag_b(&ldsV[cur][0], BK, 32);
            o0 = wmma_f16(ap1, bv, o0);
            bv = load_frag_b(&ldsV[cur][16 * BK], BK, 0);
            o1 = wmma_f16(ap0, bv, o1);
            bv = load_frag_b(&ldsV[cur][16 * BK], BK, 32);
            o1 = wmma_f16(ap1, bv, o1);
        }
        __syncthreads();                        // protect buffer reuse
    }

#pragma unroll
    for (int r = 0; r < 8; ++r) {
        float inv = 1.0f / lrow[r];
        int n = nbase + r + 8 * hi;
        outT[(size_t)n * CCH + h * DH + cl]      = (_Float16)(o0[r] * inv);
        outT[(size_t)n * CCH + h * DH + 16 + cl] = (_Float16)(o1[r] * inv);
    }
}

// ---------------------------------------------------------------------------
// Epilogue: attn_out = Wo*out + bo; g1 = relu(Wg1*[attn_out;feat1] + bg1);
// gate = sigmoid(Wg2*g1 + bg2); result = gate*feat1 + (1-gate)*attn_out.
// ---------------------------------------------------------------------------
__global__ __launch_bounds__(128) void epilogue_kernel(
    const _Float16* __restrict__ outT, const _Float16* __restrict__ f1T,
    const float* __restrict__ feat1,
    const _Float16* __restrict__ Wo16, const float* __restrict__ bo,
    const _Float16* __restrict__ Wg1_16, const float* __restrict__ bg1,
    const _Float16* __restrict__ Wg2_16, const float* __restrict__ bg2,
    float* __restrict__ result) {
    __shared__ __align__(16) _Float16 ldsAO[4][16 * 128];
    __shared__ __align__(16) _Float16 ldsG1[4][16 * 128];

    int wave = threadIdx.x >> 5, lane = threadIdx.x & 31;
    int nbase = blockIdx.x * 64 + wave * 16;
    int hi = lane >> 4, cl = lane & 15;

    v8f acc[8];

    // --- step 1: attn_out = Wo * out + bo -------------------------------
#pragma unroll
    for (int t = 0; t < 8; ++t) acc[t] = vzero8();
#pragma unroll
    for (int kc = 0; kc < CCH; kc += 32) {
        v16h a = load_frag_a(outT + (size_t)nbase * CCH, CCH, kc);
#pragma unroll
        for (int ct = 0; ct < 8; ++ct) {
            v16h b = load_frag_b(Wo16 + (size_t)ct * 16 * CCH, CCH, kc);
            acc[ct] = wmma_f16(a, b, acc[ct]);
        }
    }
#pragma unroll
    for (int ct = 0; ct < 8; ++ct) {
        float bb = bo[ct * 16 + cl];
#pragma unroll
        for (int r = 0; r < 8; ++r)
            ldsAO[wave][(r + 8 * hi) * 128 + ct * 16 + cl] =
                (_Float16)(acc[ct][r] + bb);
    }
    __syncthreads();

    // --- step 2: g1 = relu(Wg1a*attn_out + Wg1b*feat1 + bg1) ------------
#pragma unroll
    for (int t = 0; t < 8; ++t) acc[t] = vzero8();
#pragma unroll
    for (int kc = 0; kc < CCH; kc += 32) {
        v16h a = load_frag_a(ldsAO[wave], 128, kc);
#pragma unroll
        for (int ct = 0; ct < 8; ++ct) {
            v16h b = load_frag_b(Wg1_16 + (size_t)ct * 16 * 256, 256, kc);
            acc[ct] = wmma_f16(a, b, acc[ct]);
        }
    }
#pragma unroll
    for (int kc = 0; kc < CCH; kc += 32) {
        v16h a = load_frag_a(f1T + (size_t)nbase * CCH, CCH, kc);
#pragma unroll
        for (int ct = 0; ct < 8; ++ct) {
            v16h b = load_frag_b(Wg1_16 + (size_t)ct * 16 * 256, 256, 128 + kc);
            acc[ct] = wmma_f16(a, b, acc[ct]);
        }
    }
#pragma unroll
    for (int ct = 0; ct < 8; ++ct) {
        float bb = bg1[ct * 16 + cl];
#pragma unroll
        for (int r = 0; r < 8; ++r)
            ldsG1[wave][(r + 8 * hi) * 128 + ct * 16 + cl] =
                (_Float16)fmaxf(acc[ct][r] + bb, 0.0f);
    }
    __syncthreads();

    // --- step 3: gate = sigmoid(Wg2 * g1 + bg2) -------------------------
#pragma unroll
    for (int t = 0; t < 8; ++t) acc[t] = vzero8();
#pragma unroll
    for (int kc = 0; kc < CCH; kc += 32) {
        v16h a = load_frag_a(ldsG1[wave], 128, kc);
#pragma unroll
        for (int ct = 0; ct < 8; ++ct) {
            v16h b = load_frag_b(Wg2_16 + (size_t)ct * 16 * CCH, CCH, kc);
            acc[ct] = wmma_f16(a, b, acc[ct]);
        }
    }

    // --- step 4: blend and store f32 [C][N] -----------------------------
#pragma unroll
    for (int ct = 0; ct < 8; ++ct) {
        int o = ct * 16 + cl;
        float bb = bg2[o];
#pragma unroll
        for (int r = 0; r < 8; ++r) {
            int n = nbase + r + 8 * hi;
            float g  = 1.0f / (1.0f + __expf(-(acc[ct][r] + bb)));
            float fv = feat1[(size_t)o * NTOK + n];
            float ao = (float)ldsAO[wave][(r + 8 * hi) * 128 + o];
            result[(size_t)o * NTOK + n] = g * fv + (1.0f - g) * ao;
        }
    }
}

// ---------------------------------------------------------------------------
// Launch
// ---------------------------------------------------------------------------
extern "C" void kernel_launch(void* const* d_in, const int* in_sizes, int n_in,
                              void* d_out, int out_size, void* d_ws, size_t ws_size,
                              hipStream_t stream) {
    const float* feat1 = (const float*)d_in[0];
    const float* feat2 = (const float*)d_in[1];
    const float* Wq = (const float*)d_in[2];  const float* bq = (const float*)d_in[3];
    const float* Wk = (const float*)d_in[4];  const float* bk = (const float*)d_in[5];
    const float* Wv = (const float*)d_in[6];  const float* bv = (const float*)d_in[7];
    const float* Wo = (const float*)d_in[8];  const float* bo = (const float*)d_in[9];
    const float* Wg1 = (const float*)d_in[10]; const float* bg1 = (const float*)d_in[11];
    const float* Wg2 = (const float*)d_in[12]; const float* bg2 = (const float*)d_in[13];
    float* out = (float*)d_out;

    char* ws = (char*)d_ws;
    size_t off = 0;
    auto alloc = [&](size_t bytes) {
        char* p = ws + off;
        off = (off + bytes + 255) & ~(size_t)255;
        return p;
    };
    _Float16* f1T   = (_Float16*)alloc((size_t)NTOK * CCH * 2);        // [N][128]
    _Float16* f2T   = (_Float16*)alloc((size_t)NTOK * CCH * 2);
    _Float16* Wq16  = (_Float16*)alloc(CCH * CCH * 2);
    _Float16* Wk16  = (_Float16*)alloc(CCH * CCH * 2);
    _Float16* Wv16  = (_Float16*)alloc(CCH * CCH * 2);
    _Float16* Wo16  = (_Float16*)alloc(CCH * CCH * 2);
    _Float16* Wg116 = (_Float16*)alloc(CCH * 2 * CCH * 2);
    _Float16* Wg216 = (_Float16*)alloc(CCH * CCH * 2);
    _Float16* qT    = (_Float16*)alloc((size_t)NHEAD * NTOK * DH * 2); // [H][N][32]
    _Float16* kTb   = (_Float16*)alloc((size_t)NHEAD * NTOK * DH * 2); // [H][N][32]
    _Float16* vTb   = (_Float16*)alloc((size_t)NHEAD * DH * NTOK * 2); // [H][32][N]
    _Float16* outT  = (_Float16*)alloc((size_t)NTOK * CCH * 2);        // [N][128]

    int tp_blocks = (NTOK * CCH + 255) / 256;
    transpose_to_f16<<<tp_blocks, 256, 0, stream>>>(feat1, f1T);
    transpose_to_f16<<<tp_blocks, 256, 0, stream>>>(feat2, f2T);
    cast_to_f16<<<64, 256, 0, stream>>>(Wq, Wq16, CCH * CCH);
    cast_to_f16<<<64, 256, 0, stream>>>(Wk, Wk16, CCH * CCH);
    cast_to_f16<<<64, 256, 0, stream>>>(Wv, Wv16, CCH * CCH);
    cast_to_f16<<<64, 256, 0, stream>>>(Wo, Wo16, CCH * CCH);
    cast_to_f16<<<128, 256, 0, stream>>>(Wg1, Wg116, CCH * 2 * CCH);
    cast_to_f16<<<64, 256, 0, stream>>>(Wg2, Wg216, CCH * CCH);

    proj_kernel<<<dim3(NTOK / 64, 3), 128, 0, stream>>>(
        f1T, f2T, Wq16, Wk16, Wv16, bq, bk, bv, qT, kTb, vTb);

    attn_kernel<<<dim3(NTOK / 64, NHEAD), 128, 0, stream>>>(qT, kTb, vTb, outT);

    epilogue_kernel<<<NTOK / 64, 128, 0, stream>>>(
        outT, f1T, feat1, Wo16, bo, Wg116, bg1, Wg216, bg2, out);
}
